// NotebookTransformerBlock_67662914781414
// MI455X (gfx1250) — matrix operand order repaired and checked by reading, not compiled, not run on hardware
//
#include <hip/hip_runtime.h>
#include <hip/hip_bf16.h>

// ---------------- model dims ----------------
constexpr int D_MODEL   = 1024;
constexpr int N_HEADS   = 16;
constexpr int HEAD_DIM  = 64;
constexpr int KV_LATENT = 512;
constexpr int N_EXPERTS = 4;
constexpr int D_HIDDEN  = 4096;
constexpr int BB        = 2;
constexpr int TT        = 1024;
constexpr int NTOK      = BB * TT;   // 2048

typedef __attribute__((ext_vector_type(16))) _Float16 v16h;
typedef __attribute__((ext_vector_type(8)))  float    v8f;

#define LDK 40                    // padded LDS K-stride in halves; row = 80B (16B aligned)
#define TILE_ELEMS (128 * LDK)    // one 128x32 tile (padded), in halves
#define TILE_BYTES (TILE_ELEMS * 2)

// --- WMMA fragment builders (contiguous k = 16*half + e in both A and B, wave32) ---
__device__ __forceinline__ v16h lds_a_frag(const _Float16* base, int row0, int lane) {
  const int hs = (lane >> 4) & 1;
  const int m  = lane & 15;
  v16h f;
#pragma unroll
  for (int e = 0; e < 16; ++e) {
    int k = e + 16 * hs;
    f[e] = base[(row0 + m) * LDK + k];
  }
  return f;
}
__device__ __forceinline__ v16h lds_b_frag(const _Float16* base, int col0, int lane) {
  const int hs = (lane >> 4) & 1;
  const int n  = lane & 15;
  v16h f;
#pragma unroll
  for (int e = 0; e < 16; ++e) {
    int k = e + 16 * hs;
    f[e] = base[(col0 + n) * LDK + k];
  }
  return f;
}

// async 16B copy global -> LDS (gfx1250 ASYNCcnt path, bypasses VGPRs)
__device__ __forceinline__ void async_copy_b128(uint32_t lds_addr, const void* gptr) {
  uint64_t ga = (uint64_t)(uintptr_t)gptr;
  asm volatile("global_load_async_to_lds_b128 %0, %1, off"
               :: "v"(lds_addr), "v"(ga) : "memory");
}
__device__ __forceinline__ void async_wait0() {
  asm volatile("s_wait_asynccnt 0x0" ::: "memory");
}
__device__ __forceinline__ void async_wait4() {   // allow next tile's 4 chunks in flight
  asm volatile("s_wait_asynccnt 0x4" ::: "memory");
}

// ---------------- generic WMMA GEMM:  C[M,N] = A[M,K] @ W[N,K]^T
// fp16 in, fp32 accumulate. Block tile 128x128, 8 waves (4x2), wave tile 32x64.
// Double-buffered async LDS pipeline. Optional row gather (A) / scatter-add (out).
template<bool BIAS, bool RELU, bool RESID, bool OUT16, bool GATHER, bool SCATTER>
__global__ __launch_bounds__(256) void gemm_f16wmma(
    const _Float16* __restrict__ A, const _Float16* __restrict__ W,
    float* __restrict__ C, _Float16* __restrict__ Ch,
    const float* __restrict__ bias, const float* __restrict__ resid,
    const int* __restrict__ gidx,   // token index list (GATHER / SCATTER)
    const int* __restrict__ cntp,   // dynamic row count (GATHER/SCATTER), else null
    int M, int N, int K)
{
  constexpr bool DYN = GATHER || SCATTER;
  __shared__ __align__(16) _Float16 As[2 * TILE_ELEMS];
  __shared__ __align__(16) _Float16 Bs[2 * TILE_ELEMS];
  const int tid  = threadIdx.x;
  const int lane = tid & 31;
  const int wave = tid >> 5;
  const int wm   = wave & 3;       // 4 row-groups of 32
  const int wn   = wave >> 2;      // 2 col-groups of 64
  const int bm   = blockIdx.x * 128;
  const int bn   = blockIdx.y * 128;

  const int mlim = DYN ? cntp[0] : M;
  if (DYN && bm >= mlim) return;

  const uint32_t as_u = (uint32_t)(uintptr_t)(&As[0]);  // low 32b == LDS offset
  const uint32_t bs_u = (uint32_t)(uintptr_t)(&Bs[0]);

  // per-thread tile-chunk geometry (2 A-chunks + 2 B-chunks of 16B per K-step)
  int rr[2], cc[2];
  const char* aRow[2];
  const char* wRow[2];
#pragma unroll
  for (int i = 0; i < 2; ++i) {
    int q  = tid + i * 256;        // 0..511
    rr[i]  = q >> 2;               // row 0..127
    cc[i]  = (q & 3) * 16;         // byte offset in 64B row
    int ar;
    if constexpr (GATHER) ar = gidx[min(bm + rr[i], mlim - 1)];
    else                  ar = bm + rr[i];
    aRow[i] = (const char*)(A + (size_t)ar * K);
    wRow[i] = (const char*)(W + (size_t)(bn + rr[i]) * K);
  }
  auto issue = [&](int kt) {
    const int sel = kt & 1;
    const size_t koff = (size_t)kt * 64;    // 32 halves = 64 bytes
#pragma unroll
    for (int i = 0; i < 2; ++i) {
      async_copy_b128(as_u + sel * TILE_BYTES + rr[i] * (LDK * 2) + cc[i],
                      aRow[i] + koff + cc[i]);
      async_copy_b128(bs_u + sel * TILE_BYTES + rr[i] * (LDK * 2) + cc[i],
                      wRow[i] + koff + cc[i]);
    }
  };

  const int nk = K / 32;
  v8f acc[2][4] = {};
  issue(0);
  for (int kt = 0; kt < nk; ++kt) {
    if (kt + 1 < nk) { issue(kt + 1); async_wait4(); }
    else             { async_wait0(); }
    __syncthreads();                         // tile kt fully in LDS, all waves

    const _Float16* Ac = As + (kt & 1) * TILE_ELEMS;
    const _Float16* Bc = Bs + (kt & 1) * TILE_ELEMS;
    v16h a0 = lds_a_frag(Ac, wm * 32 +  0, lane);
    v16h a1 = lds_a_frag(Ac, wm * 32 + 16, lane);
#pragma unroll
    for (int tj = 0; tj < 4; ++tj) {
      v16h b = lds_b_frag(Bc, wn * 64 + tj * 16, lane);
      acc[0][tj] = __builtin_amdgcn_wmma_f32_16x16x32_f16(false, a0, false, b, (short)0, acc[0][tj], false, false);
      acc[1][tj] = __builtin_amdgcn_wmma_f32_16x16x32_f16(false, a1, false, b, (short)0, acc[1][tj], false, false);
    }
    __syncthreads();                         // done reading buf(kt) before kt+2 overwrites
  }

  const int hs = (lane >> 4) & 1;
  const int nn = lane & 15;
#pragma unroll
  for (int ti = 0; ti < 2; ++ti)
#pragma unroll
    for (int tj = 0; tj < 4; ++tj)
#pragma unroll
      for (int v = 0; v < 8; ++v) {
        int rloc = bm + wm * 32 + ti * 16 + v + 8 * hs;
        if (DYN && rloc >= mlim) continue;
        int col = bn + wn * 64 + tj * 16 + nn;
        float val = acc[ti][tj][v];
        if constexpr (BIAS)  val += bias[col];
        if constexpr (RELU)  val = fmaxf(val, 0.f);
        if constexpr (RESID) val += resid[(size_t)rloc * N + col];
        if constexpr (SCATTER) {
          C[(size_t)gidx[rloc] * N + col] += val;    // unique row per token: no atomics
        } else if constexpr (OUT16) {
          Ch[(size_t)rloc * N + col] = (_Float16)val;
        } else {
          C[(size_t)rloc * N + col] = val;
        }
      }
}

// ---------------- flash causal attention (per-head), WMMA for QK^T and PV
__global__ __launch_bounds__(256) void mla_flash_attn(
    const float* __restrict__ Q, const float* __restrict__ Kb,
    const float* __restrict__ Vb, _Float16* __restrict__ O)
{
  __shared__ _Float16 Ks[32 * 72];        // [key][d], pad 8
  __shared__ _Float16 Vs[32 * 72];
  __shared__ _Float16 Ps[8][16 * LDK];    // per-wave P staging (C-layout -> A-layout)
  const int tid  = threadIdx.x, lane = tid & 31, wave = tid >> 5;
  const int hs   = (lane >> 4) & 1, nn = lane & 15;
  const int qt   = blockIdx.x;
  const int bh   = blockIdx.y;
  const int b    = bh / N_HEADS, h = bh % N_HEADS;
  const int qrow0 = qt * 128 + wave * 16;
  const size_t headoff = (size_t)h * HEAD_DIM;

  v16h qa[2];
#pragma unroll
  for (int dc = 0; dc < 2; ++dc) {
#pragma unroll
    for (int e = 0; e < 16; ++e) {
      int k = e + 16 * hs;
      int m = lane & 15;
      qa[dc][e] = (_Float16)Q[((size_t)(b * TT + qrow0 + m)) * D_MODEL + headoff + dc * 32 + k];
    }
  }

  v8f o[4] = {};
  float rowm[8], rowl[8];
#pragma unroll
  for (int v = 0; v < 8; ++v) { rowm[v] = -1e30f; rowl[v] = 0.f; }

  const int nkt = qt * 4 + 4;
  for (int kt = 0; kt < nkt; ++kt) {
    const int kbase = kt * 32;
#pragma unroll
    for (int i = 0; i < 4; ++i) {
      int q2 = tid + i * 256;
      int r  = q2 >> 5;
      int c  = (q2 & 31) * 2;
      float2 kv = *(const float2*)(Kb + ((size_t)(b * TT + kbase + r)) * D_MODEL + headoff + c);
      Ks[r * 72 + c] = (_Float16)kv.x; Ks[r * 72 + c + 1] = (_Float16)kv.y;
      float2 vv = *(const float2*)(Vb + ((size_t)(b * TT + kbase + r)) * D_MODEL + headoff + c);
      Vs[r * 72 + c] = (_Float16)vv.x; Vs[r * 72 + c + 1] = (_Float16)vv.y;
    }
    __syncthreads();

    v8f s[2] = {};
#pragma unroll
    for (int j = 0; j < 2; ++j) {
#pragma unroll
      for (int dc = 0; dc < 2; ++dc) {
        v16h bf;
#pragma unroll
        for (int e = 0; e < 16; ++e) {
          int kd = e + 16 * hs;
          bf[e] = Ks[(j * 16 + nn) * 72 + dc * 32 + kd];
        }
        s[j] = __builtin_amdgcn_wmma_f32_16x16x32_f16(false, qa[dc], false, bf, (short)0, s[j], false, false);
      }
    }

    float pj[2][8], mnew[8];
#pragma unroll
    for (int v = 0; v < 8; ++v) {
      int row = qrow0 + v + 8 * hs;
      float mv = -1e30f;
#pragma unroll
      for (int j = 0; j < 2; ++j) {
        int col = kbase + j * 16 + nn;
        float sv = s[j][v] * 0.125f;
        sv = (col <= row) ? sv : -1e30f;
        pj[j][v] = sv;
        mv = fmaxf(mv, sv);
      }
#pragma unroll
      for (int off = 1; off < 16; off <<= 1)
        mv = fmaxf(mv, __shfl_xor(mv, off, 32));
      mnew[v] = fmaxf(rowm[v], mv);
    }
#pragma unroll
    for (int v = 0; v < 8; ++v) {
      float scale = __expf(rowm[v] - mnew[v]);
      rowm[v] = mnew[v];
      rowl[v] *= scale;
#pragma unroll
      for (int t = 0; t < 4; ++t) o[t][v] *= scale;
      float ps = 0.f;
#pragma unroll
      for (int j = 0; j < 2; ++j) {
        float p = __expf(pj[j][v] - rowm[v]);
        pj[j][v] = p;
        ps += p;
      }
#pragma unroll
      for (int off = 1; off < 16; off <<= 1)
        ps += __shfl_xor(ps, off, 32);
      rowl[v] += ps;
    }

    _Float16* pw = &Ps[wave][0];
#pragma unroll
    for (int v = 0; v < 8; ++v)
#pragma unroll
      for (int j = 0; j < 2; ++j)
        pw[(v + 8 * hs) * LDK + j * 16 + nn] = (_Float16)pj[j][v];
    __syncthreads();
    v16h pa = lds_a_frag(pw, 0, lane);

#pragma unroll
    for (int t = 0; t < 4; ++t) {
      v16h bf;
#pragma unroll
      for (int e = 0; e < 16; ++e) {
        int kk = e + 16 * hs;
        bf[e] = Vs[kk * 72 + t * 16 + nn];
      }
      o[t] = __builtin_amdgcn_wmma_f32_16x16x32_f16(false, pa, false, bf, (short)0, o[t], false, false);
    }
    __syncthreads();
  }

#pragma unroll
  for (int t = 0; t < 4; ++t)
#pragma unroll
    for (int v = 0; v < 8; ++v) {
      int row = qrow0 + v + 8 * hs;
      int d   = t * 16 + nn;
      O[((size_t)(b * TT + row)) * D_MODEL + headoff + d] = (_Float16)(o[t][v] / rowl[v]);
    }
}

// ---------------- elementwise helpers ----------------
__global__ void cvt_f32_to_f16(const float* __restrict__ S, _Float16* __restrict__ D, int n) {
  int i = blockIdx.x * blockDim.x + threadIdx.x;
  if (i < n) D[i] = (_Float16)S[i];
}

__global__ __launch_bounds__(256) void rmsnorm_kernel(
    const float* __restrict__ X, const float* __restrict__ w, _Float16* __restrict__ Y)
{
  __shared__ float red[8];
  int row = blockIdx.x, tid = threadIdx.x;
  const float* xr = X + (size_t)row * D_MODEL;
  float vals[4]; float ss = 0.f;
#pragma unroll
  for (int i = 0; i < 4; ++i) { vals[i] = xr[tid + i * 256]; ss += vals[i] * vals[i]; }
#pragma unroll
  for (int off = 1; off < 32; off <<= 1) ss += __shfl_xor(ss, off, 32);
  if ((tid & 31) == 0) red[tid >> 5] = ss;
  __syncthreads();
  float tot = 0.f;
#pragma unroll
  for (int i = 0; i < 8; ++i) tot += red[i];
  float r = rsqrtf(tot / (float)D_MODEL + 1e-6f);
#pragma unroll
  for (int i = 0; i < 4; ++i)
    Y[(size_t)row * D_MODEL + tid + i * 256] = (_Float16)(w[tid + i * 256] * vals[i] * r);
}

__global__ __launch_bounds__(256) void layernorm_kernel(
    const float* __restrict__ X, const float* __restrict__ g,
    const float* __restrict__ bta, _Float16* __restrict__ Y)
{
  __shared__ float redm[8], redv[8];
  int row = blockIdx.x, tid = threadIdx.x;
  const float* xr = X + (size_t)row * KV_LATENT;
  float a = xr[tid], b2 = xr[tid + 256];
  float s = a + b2, ss = a * a + b2 * b2;
#pragma unroll
  for (int off = 1; off < 32; off <<= 1) { s += __shfl_xor(s, off, 32); ss += __shfl_xor(ss, off, 32); }
  if ((tid & 31) == 0) { redm[tid >> 5] = s; redv[tid >> 5] = ss; }
  __syncthreads();
  float ts = 0.f, tss = 0.f;
#pragma unroll
  for (int i = 0; i < 8; ++i) { ts += redm[i]; tss += redv[i]; }
  float mu  = ts / (float)KV_LATENT;
  float var = tss / (float)KV_LATENT - mu * mu;
  float r   = rsqrtf(var + 1e-5f);
  _Float16* yr = Y + (size_t)row * KV_LATENT;
  yr[tid]       = (_Float16)((a  - mu) * r * g[tid]       + bta[tid]);
  yr[tid + 256] = (_Float16)((b2 - mu) * r * g[tid + 256] + bta[tid + 256]);
}

__global__ void rope_kernel(float* __restrict__ X) {
  int idx = blockIdx.x * blockDim.x + threadIdx.x;
  int j  = idx & 31;
  int h  = (idx >> 5) & 15;
  int bt = idx >> 9;
  float inv = __powf(10000.f, -(float)j / 32.f);
  float ang = (float)(bt & (TT - 1)) * inv;
  float sn, cs; __sincosf(ang, &sn, &cs);
  size_t base = (size_t)bt * D_MODEL + (size_t)h * HEAD_DIM;
  float x1 = X[base + j], x2 = X[base + 32 + j];
  X[base + j]      = x1 * cs - x2 * sn;
  X[base + 32 + j] = x1 * sn + x2 * cs;
}

__global__ void zero_cnt_kernel(int* __restrict__ cnt) {
  if (threadIdx.x < N_EXPERTS) cnt[threadIdx.x] = 0;
}

// gate: argmax over 4 experts; builds per-expert compacted token lists
__global__ __launch_bounds__(256) void gate_kernel(
    const _Float16* __restrict__ X, const float* __restrict__ gw,
    const float* __restrict__ gb, int* __restrict__ cnt, int* __restrict__ idxlist)
{
  int tok  = blockIdx.x * 8 + (threadIdx.x >> 5);
  int lane = threadIdx.x & 31;
  float acc[N_EXPERTS] = {0.f, 0.f, 0.f, 0.f};
  for (int d = lane; d < D_MODEL; d += 32) {
    float xv = (float)X[(size_t)tok * D_MODEL + d];
#pragma unroll
    for (int e = 0; e < N_EXPERTS; ++e) acc[e] += xv * gw[e * D_MODEL + d];
  }
#pragma unroll
  for (int e = 0; e < N_EXPERTS; ++e)
#pragma unroll
    for (int off = 1; off < 32; off <<= 1) acc[e] += __shfl_xor(acc[e], off, 32);
  if (lane == 0) {
    int best = 0; float bv = acc[0] + gb[0];
#pragma unroll
    for (int e = 1; e < N_EXPERTS; ++e) {
      float v = acc[e] + gb[e];
      if (v > bv) { bv = v; best = e; }
    }
    int pos = atomicAdd(&cnt[best], 1);
    idxlist[best * NTOK + pos] = tok;
  }
}

__global__ void copy_kernel(const float* __restrict__ S, float* __restrict__ D, int n) {
  int i = blockIdx.x * blockDim.x + threadIdx.x;
  if (i < n) D[i] = S[i];
}

// ---------------- host orchestration ----------------
extern "C" void kernel_launch(void* const* d_in, const int* in_sizes, int n_in,
                              void* d_out, int out_size, void* d_ws, size_t ws_size,
                              hipStream_t stream) {
  const float* x    = (const float*)d_in[0];
  const float* n1w  = (const float*)d_in[1];
  const float* n2w  = (const float*)d_in[2];
  const float* Wq   = (const float*)d_in[3];
  const float* Wdkv = (const float*)d_in[4];
  const float* Wuk  = (const float*)d_in[5];
  const float* Wuv  = (const float*)d_in[6];
  const float* Wo   = (const float*)d_in[7];
  const float* lng  = (const float*)d_in[8];
  const float* lnb  = (const float*)d_in[9];
  const float* gw   = (const float*)d_in[10];
  const float* gb   = (const float*)d_in[11];
  const float* W1   = (const float*)d_in[12];
  const float* b1   = (const float*)d_in[13];
  const float* W2   = (const float*)d_in[14];
  const float* b2   = (const float*)d_in[15];
  float* out = (float*)d_out;

  char* ws = (char*)d_ws;
  size_t off = 0;
  auto alloc = [&](size_t bytes) -> void* {
    void* p = ws + off;
    off += (bytes + 255) & ~(size_t)255;
    return p;
  };
  // fp16 weight mirrors (converted once per launch; ~6us of HBM traffic)
  _Float16* Wq_h   = (_Float16*)alloc((size_t)D_MODEL * D_MODEL   * 2);
  _Float16* Wdkv_h = (_Float16*)alloc((size_t)KV_LATENT * D_MODEL * 2);
  _Float16* Wuk_h  = (_Float16*)alloc((size_t)D_MODEL * KV_LATENT * 2);
  _Float16* Wuv_h  = (_Float16*)alloc((size_t)D_MODEL * KV_LATENT * 2);
  _Float16* Wo_h   = (_Float16*)alloc((size_t)D_MODEL * D_MODEL   * 2);
  _Float16* W1_h   = (_Float16*)alloc((size_t)N_EXPERTS * D_HIDDEN * D_MODEL * 2);
  _Float16* W2_h   = (_Float16*)alloc((size_t)N_EXPERTS * D_MODEL * D_HIDDEN * 2);
  // activations
  _Float16* xn1_h    = (_Float16*)alloc((size_t)NTOK * D_MODEL   * 2);
  float*    qb       = (float*)   alloc((size_t)NTOK * D_MODEL   * 4);
  float*    kb       = (float*)   alloc((size_t)NTOK * D_MODEL   * 4);
  float*    vb       = (float*)   alloc((size_t)NTOK * D_MODEL   * 4);
  float*    latent   = (float*)   alloc((size_t)NTOK * KV_LATENT * 4);
  _Float16* latent_h = (_Float16*)alloc((size_t)NTOK * KV_LATENT * 2);
  _Float16* attno_h  = (_Float16*)alloc((size_t)NTOK * D_MODEL   * 2);
  float*    xattn    = (float*)   alloc((size_t)NTOK * D_MODEL   * 4);
  _Float16* xn2_h    = (_Float16*)alloc((size_t)NTOK * D_MODEL   * 2);
  int*      cnt      = (int*)     alloc((size_t)N_EXPERTS * 4);
  int*      idxl     = (int*)     alloc((size_t)N_EXPERTS * NTOK * 4);
  _Float16* hbuf_h   = (_Float16*)alloc((size_t)NTOK * D_HIDDEN  * 2);

  dim3 blk(256);
  auto cvt = [&](const float* s, _Float16* d, size_t n) {
    cvt_f32_to_f16<<<dim3((unsigned)((n + 255) / 256)), blk, 0, stream>>>(s, d, (int)n);
  };
  cvt(Wq,   Wq_h,   (size_t)D_MODEL * D_MODEL);
  cvt(Wdkv, Wdkv_h, (size_t)KV_LATENT * D_MODEL);
  cvt(Wuk,  Wuk_h,  (size_t)D_MODEL * KV_LATENT);
  cvt(Wuv,  Wuv_h,  (size_t)D_MODEL * KV_LATENT);
  cvt(Wo,   Wo_h,   (size_t)D_MODEL * D_MODEL);
  cvt(W1,   W1_h,   (size_t)N_EXPERTS * D_HIDDEN * D_MODEL);
  cvt(W2,   W2_h,   (size_t)N_EXPERTS * D_MODEL * D_HIDDEN);

  dim3 gq(NTOK / 128, D_MODEL / 128);     // (16, 8)
  dim3 gl(NTOK / 128, KV_LATENT / 128);   // (16, 4)
  dim3 gh(NTOK / 128, D_HIDDEN / 128);    // (16, 32)
  dim3 ga(TT / 128, BB * N_HEADS);        // (8, 32)

  rmsnorm_kernel<<<NTOK, blk, 0, stream>>>(x, n1w, xn1_h);
  gemm_f16wmma<false,false,false,false,false,false><<<gq, blk, 0, stream>>>(
      xn1_h, Wq_h, qb, nullptr, nullptr, nullptr, nullptr, nullptr, NTOK, D_MODEL, D_MODEL);
  gemm_f16wmma<false,false,false,false,false,false><<<gl, blk, 0, stream>>>(
      xn1_h, Wdkv_h, latent, nullptr, nullptr, nullptr, nullptr, nullptr, NTOK, KV_LATENT, D_MODEL);
  layernorm_kernel<<<NTOK, blk, 0, stream>>>(latent, lng, lnb, latent_h);
  gemm_f16wmma<false,false,false,false,false,false><<<gq, blk, 0, stream>>>(
      latent_h, Wuk_h, kb, nullptr, nullptr, nullptr, nullptr, nullptr, NTOK, D_MODEL, KV_LATENT);
  gemm_f16wmma<false,false,false,false,false,false><<<gq, blk, 0, stream>>>(
      latent_h, Wuv_h, vb, nullptr, nullptr, nullptr, nullptr, nullptr, NTOK, D_MODEL, KV_LATENT);
  rope_kernel<<<(NTOK * 512) / 256, blk, 0, stream>>>(qb);
  rope_kernel<<<(NTOK * 512) / 256, blk, 0, stream>>>(kb);
  mla_flash_attn<<<ga, blk, 0, stream>>>(qb, kb, vb, attno_h);
  gemm_f16wmma<false,false,true,false,false,false><<<gq, blk, 0, stream>>>(
      attno_h, Wo_h, xattn, nullptr, nullptr, x, nullptr, nullptr, NTOK, D_MODEL, D_MODEL);

  // MoE: top-1 routing (4x FLOP reduction vs dense-all-experts)
  rmsnorm_kernel<<<NTOK, blk, 0, stream>>>(xattn, n2w, xn2_h);
  zero_cnt_kernel<<<1, 32, 0, stream>>>(cnt);
  gate_kernel<<<NTOK / 8, blk, 0, stream>>>(xn2_h, gw, gb, cnt, idxl);
  copy_kernel<<<(NTOK * D_MODEL) / 256, blk, 0, stream>>>(xattn, out, NTOK * D_MODEL);
  for (int e = 0; e < N_EXPERTS; ++e) {
    // h = relu(gather(xn2) @ W1[e]^T + b1[e])   (compacted rows)
    gemm_f16wmma<true,true,false,true,true,false><<<gh, blk, 0, stream>>>(
        xn2_h, W1_h + (size_t)e * D_HIDDEN * D_MODEL, nullptr, hbuf_h,
        b1 + (size_t)e * D_HIDDEN, nullptr, idxl + (size_t)e * NTOK, cnt + e,
        NTOK, D_HIDDEN, D_MODEL);
    // out[token] += h @ W2[e]^T + b2[e]         (scatter-add, unique rows)
    gemm_f16wmma<true,false,false,false,false,true><<<gq, blk, 0, stream>>>(
        hbuf_h, W2_h + (size_t)e * D_MODEL * D_HIDDEN, out, nullptr,
        b2 + (size_t)e * D_MODEL, nullptr, idxl + (size_t)e * NTOK, cnt + e,
        NTOK, D_MODEL, D_HIDDEN);
  }
}
